// CurvatureAwareLoss_43422119362760
// MI455X (gfx1250) — matrix operand order repaired and checked by reading, not compiled, not run on hardware
//
#include <hip/hip_runtime.h>
#include <hip/hip_bf16.h>
#include <math.h>

// Problem constants (match reference)
#define BB   4096   // batch
#define DD   256    // feature dim
#define HID  128    // MLP hidden
#define NTRI 528    // 32*33/2 triangular 128x128 tile pairs

typedef __attribute__((ext_vector_type(16))) __bf16 v16bf;
typedef __attribute__((ext_vector_type(8)))  float  v8f;
typedef int v4i __attribute__((vector_size(16)));              // b128 payload type
typedef __attribute__((address_space(1))) v4i g_v4i;           // global
typedef __attribute__((address_space(3))) v4i l_v4i;           // LDS

// ---- workspace layout (bytes), ~9 MB total ----
#define WS_MEAN     1024u          // 256 f32
#define WS_SCAL     2048u          // kappa
#define WS_S        4096u          // 4096 f32 : s_i = x_i^T G x_i      -> 20480
#define WS_PART     20480u         // NTRI * 8 f32 block partials       -> 37376
#define WS_SPART    65536u         // 4 * 4096 f32 (s partials)         -> 131072
#define WS_G        131072u        // 256*256 f32                       -> 393216
#define WS_CPART    393216u        // 16 * 256 f32 (colsum partials)    -> 409600
#define WS_FHI      (1u<<20)       // 4096*256 bf16 (hi of features)
#define WS_FLO      (3u<<20)       // 4096*256 bf16 (lo of features)
#define WS_FGHI     (5u<<20)       // 4096*256 bf16 (hi of F@G)
#define WS_FGLO     (7u<<20)       // 4096*256 bf16 (lo of F@G)

// ---- CDNA5 async global->LDS path (guarded; falls back to plain LDS stores)
#if defined(__has_builtin)
# if __has_builtin(__builtin_amdgcn_global_load_async_to_lds_b128)
#  define HAVE_ASYNC_LDS 1
# endif
# if __has_builtin(__builtin_amdgcn_s_wait_asynccnt)
#  define HAVE_WAIT_ASYNC 1
# endif
#endif

__device__ __forceinline__ void copy16_g2l(const unsigned short* g, unsigned short* l) {
#if defined(HAVE_ASYNC_LDS)
  __builtin_amdgcn_global_load_async_to_lds_b128(
      (g_v4i*)(uintptr_t)g,
      (l_v4i*)(unsigned)(uintptr_t)l,
      0, 0);
#else
  *(uint4*)l = *(const uint4*)g;
#endif
}
__device__ __forceinline__ void async_wait_all() {
#if defined(HAVE_ASYNC_LDS)
# if defined(HAVE_WAIT_ASYNC)
  __builtin_amdgcn_s_wait_asynccnt(0);
# else
  asm volatile("s_wait_asynccnt 0x0" ::: "memory");
# endif
#endif
}

__device__ __forceinline__ unsigned short f2bf(float f) {
  unsigned u = __float_as_uint(f);
  u += 0x7FFFu + ((u >> 16) & 1u);          // round-to-nearest-even
  return (unsigned short)(u >> 16);
}
__device__ __forceinline__ float bf2f(unsigned short h) {
  return __uint_as_float(((unsigned)h) << 16);
}

// ---------------- column-sum partials (deterministic, no atomics) -----------
__global__ void k_colsum(const float* __restrict__ F, char* ws) {
  float* cpart = (float*)(ws + WS_CPART);
  int t = threadIdx.x;
  int r0 = blockIdx.x * 256;
  float acc = 0.0f;
  for (int r = 0; r < 256; ++r) acc += F[(size_t)(r0 + r) * DD + t];
  cpart[blockIdx.x * DD + t] = acc;
}

// ---------------- mean + curvature MLP: kappa = -softplus(tanh(m@w1+b1)@w2+b2)
__global__ void k_stats(const float* __restrict__ w1, const float* __restrict__ b1,
                        const float* __restrict__ w2, const float* __restrict__ b2,
                        char* ws) {
  __shared__ float mean_s[DD];
  __shared__ float hv[HID];
  const float* cpart = (const float*)(ws + WS_CPART);
  float* mean = (float*)(ws + WS_MEAN);
  float* scal = (float*)(ws + WS_SCAL);
  int t = threadIdx.x;
  float cs = 0.0f;
  #pragma unroll
  for (int b = 0; b < 16; ++b) cs += cpart[b * DD + t];
  float m = cs * (1.0f / (float)BB);
  mean[t] = m; mean_s[t] = m;
  __syncthreads();
  if (t < HID) {
    float acc = b1[t];
    for (int j = 0; j < DD; ++j) acc = fmaf(mean_s[j], w1[j * HID + t], acc);
    hv[t] = tanhf(acc) * w2[t];
  }
  __syncthreads();
  for (int st = HID / 2; st > 0; st >>= 1) {
    if (t < st) hv[t] += hv[t + st];
    __syncthreads();
  }
  if (t == 0) {
    float z = hv[0] + b2[0];
    float sp = (z > 20.0f) ? z : log1pf(expf(z));
    scal[0] = -sp;
  }
}

// ---------------- features -> bf16 hi/lo split ------------------------------
__global__ void k_convF(const float* __restrict__ F, char* ws) {
  unsigned short* fh = (unsigned short*)(ws + WS_FHI);
  unsigned short* fl = (unsigned short*)(ws + WS_FLO);
  size_t idx = (size_t)blockIdx.x * 256 + threadIdx.x;
  float v = F[idx];
  unsigned short h = f2bf(v);
  fh[idx] = h;
  fl[idx] = f2bf(v - bf2f(h));
}

// ---------------- G = I + fc^T fc / B  (deterministic, 256 blocks) ----------
__global__ void __launch_bounds__(256) k_gemmG(const float* __restrict__ F, char* ws) {
  const float* mean = (const float*)(ws + WS_MEAN);
  float* G = (float*)(ws + WS_G);
  __shared__ float Am[64][17];
  __shared__ float Bn[64][17];
  int t = threadIdx.x, tx = t & 15, ty = t >> 4;
  int m0 = blockIdx.x * 16, n0 = blockIdx.y * 16;
  float acc = 0.0f;
  for (int k0 = 0; k0 < BB; k0 += 64) {
    __syncthreads();
    #pragma unroll
    for (int r = 0; r < 4; ++r) {
      int idx = t + r * 256;
      int kk = idx >> 4, cc = idx & 15;
      Am[kk][cc] = F[(size_t)(k0 + kk) * DD + m0 + cc] - mean[m0 + cc];
      Bn[kk][cc] = F[(size_t)(k0 + kk) * DD + n0 + cc] - mean[n0 + cc];
    }
    __syncthreads();
    #pragma unroll
    for (int k = 0; k < 64; ++k) acc = fmaf(Am[k][ty], Bn[k][tx], acc);
  }
  int m = m0 + ty, n = n0 + tx;
  G[m * DD + n] = acc * (1.0f / (float)BB) + (m == n ? 1.0f : 0.0f);
}

// ---------------- FG = F @ G -> bf16 hi/lo + s partials ---------------------
__global__ void __launch_bounds__(256) k_gemmFG(const float* __restrict__ F, char* ws) {
  const float* G = (const float*)(ws + WS_G);
  float* spart = (float*)(ws + WS_SPART);
  unsigned short* fgh = (unsigned short*)(ws + WS_FGHI);
  unsigned short* fgl = (unsigned short*)(ws + WS_FGLO);
  __shared__ float Ft[64][65];
  __shared__ float Gt[64][65];
  int t = threadIdx.x, tx = t & 15, ty = t >> 4;
  int i0 = blockIdx.x * 64, n0 = blockIdx.y * 64;
  float acc[4][4] = {};
  for (int k0 = 0; k0 < DD; k0 += 64) {
    __syncthreads();
    #pragma unroll
    for (int r = 0; r < 16; ++r) {
      int idx = t + r * 256;
      int rr = idx >> 6, cc = idx & 63;
      Ft[rr][cc] = F[(size_t)(i0 + rr) * DD + k0 + cc];
      Gt[rr][cc] = G[(k0 + rr) * DD + n0 + cc];
    }
    __syncthreads();
    for (int k = 0; k < 64; ++k) {
      float av[4], bv[4];
      #pragma unroll
      for (int a = 0; a < 4; ++a) av[a] = Ft[ty * 4 + a][k];
      #pragma unroll
      for (int b = 0; b < 4; ++b) bv[b] = Gt[k][tx * 4 + b];
      #pragma unroll
      for (int a = 0; a < 4; ++a)
        #pragma unroll
        for (int b = 0; b < 4; ++b) acc[a][b] = fmaf(av[a], bv[b], acc[a][b]);
    }
  }
  float sp[4] = {0.f, 0.f, 0.f, 0.f};
  #pragma unroll
  for (int a = 0; a < 4; ++a) {
    int i = i0 + ty * 4 + a;
    #pragma unroll
    for (int b = 0; b < 4; ++b) {
      int n = n0 + tx * 4 + b;
      float v = acc[a][b];
      unsigned short h = f2bf(v);
      fgh[(size_t)i * DD + n] = h;
      fgl[(size_t)i * DD + n] = f2bf(v - bf2f(h));
      sp[a] = fmaf(v, F[(size_t)i * DD + n], sp[a]);
    }
  }
  __syncthreads();
  float* red = &Ft[0][0];   // reuse LDS: 64 x 16
  #pragma unroll
  for (int a = 0; a < 4; ++a) red[(ty * 4 + a) * 16 + tx] = sp[a];
  __syncthreads();
  if (t < 64) {
    float tot = 0.f;
    #pragma unroll
    for (int x = 0; x < 16; ++x) tot += red[t * 16 + x];
    spart[blockIdx.y * BB + i0 + t] = tot;   // deterministic, no atomics
  }
}

// ---------------- s_i = sum of 4 partials -----------------------------------
__global__ void k_s(char* ws) {
  const float* spart = (const float*)(ws + WS_SPART);
  float* s = (float*)(ws + WS_S);
  int i = blockIdx.x * 256 + threadIdx.x;
  s[i] = ((spart[i] + spart[BB + i]) + spart[2 * BB + i]) + spart[3 * BB + i];
}

// ---------------- fused pairwise WMMA kernel (triangular tiles) -------------
// P = FG @ F^T via bf16 hi/lo compensated WMMA (3 products, f32 accum).
// LDS-staged tiles (async global->LDS), epilogue: d2, masked exp, min.
#define LSTR 40   // shorts per LDS row: 32 data + 8 pad (80B, 16B aligned, conflict-free)
__global__ void __launch_bounds__(256) k_pair(const int* __restrict__ labels, char* ws) {
  __shared__ unsigned short sAhi[128 * LSTR];
  __shared__ unsigned short sAlo[128 * LSTR];
  __shared__ unsigned short sBhi[128 * LSTR];
  __shared__ unsigned short sBlo[128 * LSTR];
  __shared__ float rP[256], rN[256], rM[256], rPC[256], rNC[256];

  const unsigned short* fgh = (const unsigned short*)(ws + WS_FGHI);
  const unsigned short* fgl = (const unsigned short*)(ws + WS_FGLO);
  const unsigned short* fh  = (const unsigned short*)(ws + WS_FHI);
  const unsigned short* fl  = (const unsigned short*)(ws + WS_FLO);
  const float* s = (const float*)(ws + WS_S);
  float* part = (float*)(ws + WS_PART);
  const float kappa = ((const float*)(ws + WS_SCAL))[0];

  // triangular decode: x = bj*(bj+1)/2 + bi, bi <= bj
  int x = blockIdx.x;
  int bj = 0;
  while ((((bj + 1) * (bj + 2)) >> 1) <= x) ++bj;
  int bi = x - ((bj * (bj + 1)) >> 1);

  int t = threadIdx.x;
  int lane = t & 31, wave = t >> 5;
  int waveRow = wave >> 1, waveCol = wave & 1;
  int half = lane >> 4, l16 = lane & 15;

  int rowBlk = bi * 128, colBlk = bj * 128;
  int rowBase = waveRow * 32;   // local
  int colBase = waveCol * 64;   // local

  union Frag { uint4 q[2]; v16bf v; };
  v8f c[2][4] = {};

  for (int k0 = 0; k0 < DD; k0 += 32) {
    __syncthreads();            // previous iteration fully consumed
    #pragma unroll
    for (int p = 0; p < 8; ++p) {
      int id = t + (p & 1) * 256;           // 0..511
      int row = id >> 2, seg = id & 3;      // 128 rows x 4 x 16B
      const unsigned short* gsrc; unsigned short* ldst; int rb;
      if (p < 2)      { gsrc = fgh; ldst = sAhi; rb = rowBlk; }
      else if (p < 4) { gsrc = fgl; ldst = sAlo; rb = rowBlk; }
      else if (p < 6) { gsrc = fh;  ldst = sBhi; rb = colBlk; }
      else            { gsrc = fl;  ldst = sBlo; rb = colBlk; }
      copy16_g2l(gsrc + (size_t)(rb + row) * DD + k0 + seg * 8,
                 ldst + row * LSTR + seg * 8);
    }
    async_wait_all();
    __syncthreads();            // tiles visible to all waves

    Frag ah[2], al[2], bh[4], bl[4];
    #pragma unroll
    for (int tm = 0; tm < 2; ++tm) {
      // A frag: element e -> K = 16*(e>=8) + 8*half + e%8
      int ro = (rowBase + tm * 16 + l16) * LSTR + half * 8;
      ah[tm].q[0] = *(const uint4*)(sAhi + ro);
      ah[tm].q[1] = *(const uint4*)(sAhi + ro + 16);
      al[tm].q[0] = *(const uint4*)(sAlo + ro);
      al[tm].q[1] = *(const uint4*)(sAlo + ro + 16);
    }
    #pragma unroll
    for (int tn = 0; tn < 4; ++tn) {
      // B frag: element e -> K = 16*(e>=8) + 8*half + e%8
      int ro = (colBase + tn * 16 + l16) * LSTR + half * 8;
      bh[tn].q[0] = *(const uint4*)(sBhi + ro);
      bh[tn].q[1] = *(const uint4*)(sBhi + ro + 16);
      bl[tn].q[0] = *(const uint4*)(sBlo + ro);
      bl[tn].q[1] = *(const uint4*)(sBlo + ro + 16);
    }
    #pragma unroll
    for (int tm = 0; tm < 2; ++tm)
      #pragma unroll
      for (int tn = 0; tn < 4; ++tn) {
        c[tm][tn] = __builtin_amdgcn_wmma_f32_16x16x32_bf16(
            false, ah[tm].v, false, bh[tn].v, (short)0, c[tm][tn], false, false);
        c[tm][tn] = __builtin_amdgcn_wmma_f32_16x16x32_bf16(
            false, ah[tm].v, false, bl[tn].v, (short)0, c[tm][tn], false, false);
        c[tm][tn] = __builtin_amdgcn_wmma_f32_16x16x32_bf16(
            false, al[tm].v, false, bh[tn].v, (short)0, c[tm][tn], false, false);
      }
  }

  // epilogue: C layout -> VGPR r: M = r + 8*half, N = l16
  float pos = 0.f, neg = 0.f, mind2 = __uint_as_float(0x7F800000u);
  float pcnt = 0.f, ncnt = 0.f;
  #pragma unroll
  for (int tn = 0; tn < 4; ++tn) {
    int j = colBlk + colBase + tn * 16 + l16;
    float sj = s[j];
    int lj = labels[j];
    #pragma unroll
    for (int tm = 0; tm < 2; ++tm) {
      int ibase = rowBlk + rowBase + tm * 16 + half * 8;
      #pragma unroll
      for (int r = 0; r < 8; ++r) {
        int i = ibase + r;
        if (i == j) continue;
        float p = c[tm][tn][r];
        float d2 = fmaxf(s[i] + sj - 2.0f * p, 0.0f);
        if (labels[i] == lj) { pos += __expf(kappa * d2); pcnt += 1.0f; }
        else { neg += __expf(-kappa * d2); ncnt += 1.0f; mind2 = fminf(mind2, d2); }
      }
    }
  }

  rP[t] = pos; rN[t] = neg; rM[t] = mind2; rPC[t] = pcnt; rNC[t] = ncnt;
  __syncthreads();
  for (int st = 128; st > 0; st >>= 1) {
    if (t < st) {
      rP[t] += rP[t + st]; rN[t] += rN[t + st];
      rM[t] = fminf(rM[t], rM[t + st]);
      rPC[t] += rPC[t + st]; rNC[t] += rNC[t + st];
    }
    __syncthreads();
  }
  if (t == 0) {
    float f = (bi == bj) ? 1.0f : 2.0f;     // symmetry doubling
    float* pp = part + (size_t)blockIdx.x * 8;
    pp[0] = rP[0] * f; pp[1] = rN[0] * f; pp[2] = rM[0];
    pp[3] = rPC[0] * f; pp[4] = rNC[0] * f;
  }
}

// ---------------- final deterministic reduction + scalar combine ------------
__global__ void k_final(const float* __restrict__ kappa_param, char* ws, float* out) {
  __shared__ float rP[256], rN[256], rM[256], rPC[256], rNC[256];
  const float* part = (const float*)(ws + WS_PART);
  int t = threadIdx.x;
  float pos = 0.f, neg = 0.f, mind2 = __uint_as_float(0x7F800000u);
  float pc = 0.f, nc = 0.f;
  for (int b = t; b < NTRI; b += 256) {
    const float* pp = part + (size_t)b * 8;
    pos += pp[0]; neg += pp[1]; mind2 = fminf(mind2, pp[2]);
    pc += pp[3]; nc += pp[4];
  }
  rP[t] = pos; rN[t] = neg; rM[t] = mind2; rPC[t] = pc; rNC[t] = nc;
  __syncthreads();
  for (int st = 128; st > 0; st >>= 1) {
    if (t < st) {
      rP[t] += rP[t + st]; rN[t] += rN[t + st];
      rM[t] = fminf(rM[t], rM[t + st]);
      rPC[t] += rPC[t + st]; rNC[t] += rNC[t + st];
    }
    __syncthreads();
  }
  if (t == 0) {
    float posL = rP[0] / fmaxf(rPC[0], 1.0f);
    float negL = rN[0] / fmaxf(rNC[0], 1.0f);
    float contrastive = posL - negL + 1.0f;           // MARGIN = 1
    float dist = sqrtf(fmaxf(rM[0], 1e-12f));
    float delta = fmaxf(dist, 0.1f);
    float reg = 0.1f * fmaxf(0.0f, kappa_param[0] + 1.0f / (delta * delta));
    out[0] = contrastive + reg;
  }
}

extern "C" void kernel_launch(void* const* d_in, const int* in_sizes, int n_in,
                              void* d_out, int out_size, void* d_ws, size_t ws_size,
                              hipStream_t stream) {
  const float* F      = (const float*)d_in[0];
  const float* w1     = (const float*)d_in[1];
  const float* b1     = (const float*)d_in[2];
  const float* w2     = (const float*)d_in[3];
  const float* b2     = (const float*)d_in[4];
  const float* kparam = (const float*)d_in[5];
  const int*   labels = (const int*)d_in[6];
  char*  ws  = (char*)d_ws;     // needs ~9.5 MB
  float* out = (float*)d_out;

  k_colsum<<<16, 256, 0, stream>>>(F, ws);
  k_stats <<<1, 256, 0, stream>>>(w1, b1, w2, b2, ws);
  k_convF <<<(BB * DD) / 256, 256, 0, stream>>>(F, ws);
  k_gemmG <<<dim3(16, 16), 256, 0, stream>>>(F, ws);
  k_gemmFG<<<dim3(64, 4), 256, 0, stream>>>(F, ws);
  k_s     <<<BB / 256, 256, 0, stream>>>(ws);
  k_pair  <<<NTRI, 256, 0, stream>>>(labels, ws);
  k_final <<<1, 256, 0, stream>>>(kparam, ws, out);
}
